// SelfAttention_68504728371650
// MI455X (gfx1250) — matrix operand order repaired and checked by reading, not compiled, not run on hardware
//
#include <hip/hip_runtime.h>
#include <hip/hip_bf16.h>
#include <climits>

// ---------------- problem constants (from reference) ----------------
#define NB   4
#define LSEQ 2048
#define DM   1024
#define NH   16
#define DH   64
#define NL   (NB * LSEQ)   // 8192 rows

typedef __bf16 bf16_t;
typedef bf16_t bf16x8 __attribute__((ext_vector_type(8)));
typedef bf16_t v16bf  __attribute__((ext_vector_type(16)));
typedef float  v8f    __attribute__((ext_vector_type(8)));

// -------------------------------------------------------------------
// A-tile loader: 16x32 bf16 A-matrix, row-major source (lda elements).
// Contiguous k order used here: lanes 0-15 -> M=0..15, elem e -> K = k0 + hi*16 + e.
// Two b128 loads per lane.
// -------------------------------------------------------------------
__device__ __forceinline__ v16bf load_a_tile(const bf16_t* __restrict__ base,
                                             int lda, int k0, int lane) {
    const int m  = lane & 15;
    const int hi = lane >> 4;
    const bf16_t* p = base + (size_t)m * lda + k0 + hi * 16;
    bf16x8 lo = *(const bf16x8*)(p);
    bf16x8 hh = *(const bf16x8*)(p + 8);
    v16bf a;
#pragma unroll
    for (int i = 0; i < 8; ++i) { a[i] = lo[i]; a[8 + i] = hh[i]; }
    return a;
}

// -------------------------------------------------------------------
// B-tile loader: 32x16 bf16 B-matrix. Source is stored "Bt" = row-major
// [N][K] (i.e. B transposed), so each lane reads 16 contiguous bf16.
// Contiguous k order used here: N = lane&15, elem e -> K = k0 + hi*16 + e.
// -------------------------------------------------------------------
__device__ __forceinline__ v16bf load_bt_tile(const bf16_t* __restrict__ bt,
                                              int ldb, int k0, int lane) {
    const int n  = lane & 15;
    const int hi = lane >> 4;
    const bf16_t* p = bt + (size_t)n * ldb + k0 + hi * 16;
    bf16x8 lo = *(const bf16x8*)(p);
    bf16x8 hh = *(const bf16x8*)(p + 8);
    v16bf b;
#pragma unroll
    for (int i = 0; i < 8; ++i) { b[i] = lo[i]; b[8 + i] = hh[i]; }
    return b;
}

// -------------------------------------------------------------------
// Elementwise converters
// -------------------------------------------------------------------
__global__ void k_f32_to_bf16(const float* __restrict__ in,
                              bf16_t* __restrict__ out, int n) {
    int i = blockIdx.x * blockDim.x + threadIdx.x;
    if (i < n) out[i] = (bf16_t)in[i];
}

// out[c*rows + r] = in[r*cols + c]  (store W as [N][K] for B-operand loads)
__global__ void k_transpose_bf16(const float* __restrict__ in,
                                 bf16_t* __restrict__ out, int rows, int cols) {
    int i = blockIdx.x * blockDim.x + threadIdx.x;
    if (i < rows * cols) {
        int r = i / cols, c = i % cols;
        out[(size_t)c * rows + r] = (bf16_t)in[i];
    }
}

// ybar[token, h*64+d] = bf16( w[n,h,tok] * V[token, h*64+d] )
__global__ void k_scale_v(const bf16_t* __restrict__ V,
                          const float* __restrict__ w,
                          bf16_t* __restrict__ Y) {
    int i = blockIdx.x * blockDim.x + threadIdx.x;
    if (i < NL * DM) {
        int tok = i / DM;
        int hh  = (i % DM) / DH;
        int nb  = tok / LSEQ;
        int lt  = tok % LSEQ;
        float wv = w[((size_t)nb * NH + hh) * LSEQ + lt];
        Y[i] = (bf16_t)(wv * (float)V[i]);
    }
}

// -------------------------------------------------------------------
// Tiled WMMA GEMM:  C[M,N] = A[M,K] * B[K,N], B given transposed [N][K].
// 128 threads = 4 waves; block computes 128(M) x 64(N); each wave owns
// 32 rows x 64 cols = 2x4 accumulator tiles (64 acc VGPRs). Per K-step
// (32): 4 A-loads + 8 B-loads (b128) feed 8 WMMAs -> 1.5 VMEM/WMMA.
// B panels are shared by all 4 waves via L0/L2. Software prefetch one
// K-step ahead (global_prefetch_b8).
// -------------------------------------------------------------------
template <typename OT>
__global__ __launch_bounds__(128)
void k_gemm_nt_bf16(const bf16_t* __restrict__ A,
                    const bf16_t* __restrict__ Bt,
                    OT* __restrict__ C, int M, int N, int K) {
    const int lane = threadIdx.x & 31;
    const int wave = threadIdx.x >> 5;
    const int m0   = blockIdx.x * 128 + wave * 32;
    const int n0b  = blockIdx.y * 64;
    const int col  = lane & 15;
    const int rowb = (lane >> 4) * 8;

    v8f acc[2][4] = {};
    const bf16_t* Arow0 = A + (size_t)m0 * K;
    const bf16_t* Arow1 = A + (size_t)(m0 + 16) * K;

    for (int k0 = 0; k0 < K; k0 += 32) {
        if (k0 + 32 < K) {  // prefetch next K-panel (A rows + B rows)
            __builtin_prefetch(Arow0 + (size_t)col * K + k0 + 32, 0, 0);
            __builtin_prefetch(Bt + (size_t)(n0b + lane) * K + k0 + 32, 0, 0);
        }
        v16bf a0 = load_a_tile(Arow0, K, k0, lane);
        v16bf a1 = load_a_tile(Arow1, K, k0, lane);
#pragma unroll
        for (int t = 0; t < 4; ++t) {
            v16bf b = load_bt_tile(Bt + (size_t)(n0b + t * 16) * K, K, k0, lane);
            acc[0][t] = __builtin_amdgcn_wmma_f32_16x16x32_bf16(
                false, a0, false, b, (short)0, acc[0][t], false, false);
            acc[1][t] = __builtin_amdgcn_wmma_f32_16x16x32_bf16(
                false, a1, false, b, (short)0, acc[1][t], false, false);
        }
    }
#pragma unroll
    for (int g = 0; g < 2; ++g)
#pragma unroll
        for (int t = 0; t < 4; ++t)
#pragma unroll
            for (int j = 0; j < 8; ++j)
                C[(size_t)(m0 + g * 16 + rowb + j) * N + n0b + t * 16 + col] =
                    (OT)acc[g][t][j];
}

// -------------------------------------------------------------------
// Score kernel: one wave per (batch nb, head hh, 16-query tile q0).
// C_tile[16x16] = Q_tile(16x64) * K_tile(64x16) via 2 WMMAs; keep a
// per-lane running (max, argmax) over all key columns, capture the
// diagonal tile, then a 16-lane LDS reduction produces
//   w[nb,hh,q] = (argmax_k C[q,k] == q) ? C[q,q] : 0
// First-occurrence argmax semantics: strictly-greater update per lane
// (column index only increases), min-index tie-break in the reduction.
// -------------------------------------------------------------------
__global__ __launch_bounds__(32)
void k_score_argmax(const bf16_t* __restrict__ Q,
                    const bf16_t* __restrict__ Kmat,
                    const int* __restrict__ mask,
                    float* __restrict__ w) {
    const int lane = threadIdx.x;
    const int q0   = blockIdx.x * 16;
    const int hh   = blockIdx.y;
    const int nb   = blockIdx.z;
    const int col  = lane & 15;
    const int rowb = (lane >> 4) * 8;

    const bf16_t* Abase = Q + ((size_t)(nb * LSEQ + q0)) * DM + hh * DH;
    v16bf a0 = load_a_tile(Abase, DM, 0, lane);
    v16bf a1 = load_a_tile(Abase, DM, 32, lane);

    float runmax[8];
    int   runidx[8];
#pragma unroll
    for (int j = 0; j < 8; ++j) { runmax[j] = -__builtin_inff(); runidx[j] = INT_MAX; }
    float diagv = 0.0f;

    for (int kt = 0; kt < LSEQ / 16; ++kt) {
        const int n0 = kt * 16;
        const bf16_t* Bbase = Kmat + ((size_t)(nb * LSEQ + n0)) * DM + hh * DH;
        v16bf b0 = load_bt_tile(Bbase, DM, 0, lane);
        v16bf b1 = load_bt_tile(Bbase, DM, 32, lane);
        v8f accv = {};
        accv = __builtin_amdgcn_wmma_f32_16x16x32_bf16(false, a0, false, b0,
                                                       (short)0, accv, false, false);
        accv = __builtin_amdgcn_wmma_f32_16x16x32_bf16(false, a1, false, b1,
                                                       (short)0, accv, false, false);
        const int mv = mask[(size_t)nb * LSEQ + n0 + col];
        if (mv != 0) {
#pragma unroll
            for (int j = 0; j < 8; ++j)
                if (accv[j] > runmax[j]) { runmax[j] = accv[j]; runidx[j] = n0 + col; }
        }
        if (n0 == q0) {  // diagonal tile: element with N==M lives where col==rowb+j
#pragma unroll
            for (int j = 0; j < 8; ++j)
                if (col - rowb == j) diagv = accv[j];
        }
    }

    __shared__ float smax[8][32];
    __shared__ int   sidx[8][32];
    __shared__ float sdiag[16];
#pragma unroll
    for (int j = 0; j < 8; ++j) { smax[j][lane] = runmax[j]; sidx[j][lane] = runidx[j]; }
    {
        int j = col - rowb;
        if (j >= 0 && j < 8) sdiag[col] = diagv;  // diagonal row == col
    }
    __syncthreads();

    if (lane < 16) {
        const int r = lane;
        const int j = r & 7;
        const int off = (r < 8) ? 0 : 16;
        float best = -__builtin_inff();
        int bidx = INT_MAX;
#pragma unroll
        for (int c = 0; c < 16; ++c) {
            float v  = smax[j][off + c];
            int   id = sidx[j][off + c];
            if (v > best || (v == best && id < bidx)) { best = v; bidx = id; }
        }
        const int q = q0 + r;
        w[((size_t)nb * NH + hh) * LSEQ + q] = (bidx == q) ? sdiag[r] : 0.0f;
    }
}

// -------------------------------------------------------------------
// Host-side orchestration
// -------------------------------------------------------------------
extern "C" void kernel_launch(void* const* d_in, const int* in_sizes, int n_in,
                              void* d_out, int out_size, void* d_ws, size_t ws_size,
                              hipStream_t stream) {
    const float* Z    = (const float*)d_in[0];
    const int*   mask = (const int*)d_in[1];
    const float* Wv   = (const float*)d_in[2];
    const float* Wk   = (const float*)d_in[3];
    const float* Wq   = (const float*)d_in[4];
    const float* Wo   = (const float*)d_in[5];
    float* out = (float*)d_out;

    // workspace layout (bytes)
    char* ws = (char*)d_ws;
    size_t off = 0;
    bf16_t* Zb  = (bf16_t*)(ws + off); off += (size_t)NL * DM * 2;   // 16 MiB
    bf16_t* Wvt = (bf16_t*)(ws + off); off += (size_t)DM * DM * 2;   //  2 MiB
    bf16_t* Wkt = (bf16_t*)(ws + off); off += (size_t)DM * DM * 2;
    bf16_t* Wqt = (bf16_t*)(ws + off); off += (size_t)DM * DM * 2;
    bf16_t* Wot = (bf16_t*)(ws + off); off += (size_t)DM * DM * 2;
    bf16_t* Qb  = (bf16_t*)(ws + off); off += (size_t)NL * DM * 2;   // 16 MiB
    bf16_t* Kb  = (bf16_t*)(ws + off); off += (size_t)NL * DM * 2;
    bf16_t* Vb  = (bf16_t*)(ws + off); off += (size_t)NL * DM * 2;
    float*  wb  = (float*)(ws + off);  off += (size_t)NB * NH * LSEQ * 4;
    bf16_t* Yb  = (bf16_t*)(ws + off); off += (size_t)NL * DM * 2;
    (void)ws_size; (void)in_sizes; (void)n_in; (void)out_size;

    const int ETH = 256;
    // 1) convert Z; transpose+convert weights
    k_f32_to_bf16<<<(NL * DM + ETH - 1) / ETH, ETH, 0, stream>>>(Z, Zb, NL * DM);
    k_transpose_bf16<<<(DM * DM + ETH - 1) / ETH, ETH, 0, stream>>>(Wv, Wvt, DM, DM);
    k_transpose_bf16<<<(DM * DM + ETH - 1) / ETH, ETH, 0, stream>>>(Wk, Wkt, DM, DM);
    k_transpose_bf16<<<(DM * DM + ETH - 1) / ETH, ETH, 0, stream>>>(Wq, Wqt, DM, DM);
    k_transpose_bf16<<<(DM * DM + ETH - 1) / ETH, ETH, 0, stream>>>(Wo, Wot, DM, DM);

    // 2) projections  Q/K/V = Z @ W   (block tile 128x64)
    dim3 ggrid(NL / 128, DM / 64);
    k_gemm_nt_bf16<bf16_t><<<ggrid, 128, 0, stream>>>(Zb, Wqt, Qb, NL, DM, DM);
    k_gemm_nt_bf16<bf16_t><<<ggrid, 128, 0, stream>>>(Zb, Wkt, Kb, NL, DM, DM);
    k_gemm_nt_bf16<bf16_t><<<ggrid, 128, 0, stream>>>(Zb, Wvt, Vb, NL, DM, DM);

    // 3) hard-argmax attention weights
    dim3 sgrid(LSEQ / 16, NH, NB);
    k_score_argmax<<<sgrid, 32, 0, stream>>>(Qb, Kb, mask, wb);

    // 4) ybar = w * V
    k_scale_v<<<(NL * DM + ETH - 1) / ETH, ETH, 0, stream>>>(Vb, wb, Yb);

    // 5) out = ybar @ Wo  (f32 output)
    k_gemm_nt_bf16<float><<<ggrid, 128, 0, stream>>>(Yb, Wot, out, NL, DM, DM);
}
